// MultiHeadAttention_16810501997027
// MI455X (gfx1250) — compile-verified, matrix operand order checked by reading, not run on hardware
//
#include <hip/hip_runtime.h>
#include <hip/hip_bf16.h>

#define D_EMB 1024
#define NH    16
#define DH    64
#define SEQ   2048
#define BATCH 2
#define M_TOT (BATCH * SEQ)   // 4096

typedef __attribute__((ext_vector_type(16))) __bf16 v16bf;
typedef __attribute__((ext_vector_type(8)))  float  v8f;

union Frag {
    v16bf bf;
    uint4 q[2];
};

__device__ __forceinline__ unsigned short f2bf(float f) {
    __bf16 b = (__bf16)f;
    return __builtin_bit_cast(unsigned short, b);
}

// Load one 16x32 bf16 WMMA operand fragment (A layout; B uses the same
// per-lane pattern with "row" meaning the operand column, per CDNA5 ISA
// 16-bit matrix layouts).  lane L -> row (L&15); K halves:
//   lanes 0-15 : K = 0..7 and 16..23
//   lanes 16-31: K = 8..15 and 24..31
__device__ __forceinline__ Frag load_frag(const unsigned short* __restrict__ base, int ld) {
    Frag f;
    const int lane = threadIdx.x & 31;
    const int row  = lane & 15;
    const int ko   = (lane < 16) ? 0 : 8;
    const unsigned short* p = base + (size_t)row * ld + ko;
    f.q[0] = *(const uint4*)(p);
    f.q[1] = *(const uint4*)(p + 16);
    return f;
}

__device__ __forceinline__ v8f wmma_bf16(const Frag& a, const Frag& b, v8f c) {
    return __builtin_amdgcn_wmma_f32_16x16x32_bf16(
        /*neg_a=*/false, a.bf, /*neg_b=*/false, b.bf,
        /*c_mod=*/(short)0, c, /*reuse_a=*/false, /*reuse_b=*/false);
}

// ---------------------------------------------------------------------------
// fp32 -> bf16 conversion (vectorized x4)
// ---------------------------------------------------------------------------
__global__ __launch_bounds__(256) void cvt_kernel(const float* __restrict__ in,
                                                  unsigned short* __restrict__ out,
                                                  int n4) {
    int i = blockIdx.x * blockDim.x + threadIdx.x;
    if (i < n4) {
        float4 f = ((const float4*)in)[i];
        ushort4 u;
        u.x = f2bf(f.x); u.y = f2bf(f.y); u.z = f2bf(f.z); u.w = f2bf(f.w);
        ((ushort4*)out)[i] = u;
    }
}

// ---------------------------------------------------------------------------
// GEMM: out[m,n] = sum_k A[m,k] * W[n,k] + bias[n]
// A: [M,K] bf16 row-major, W: [N,K] bf16 row-major.
// mode 0: store bf16 at [b,h,s,d]   (Q / K layout)
// mode 1: store bf16 at [b,h,d,s]   (V transposed)
// mode 2: store fp32 row-major [M,N] (final output)
// 8 waves per WG: 4 along M x 2 along N; each wave = 64x64 tile.
// ---------------------------------------------------------------------------
__global__ __launch_bounds__(256)
void gemm_bias_kernel(const unsigned short* __restrict__ A,
                      const unsigned short* __restrict__ W,
                      const float* __restrict__ bias,
                      void* __restrict__ out,
                      int M, int N, int K, int mode) {
    const int wave = threadIdx.x >> 5;
    const int lane = threadIdx.x & 31;
    const int wm = wave >> 1;
    const int wn = wave & 1;
    const int m0 = blockIdx.y * 256 + wm * 64;
    const int n0 = blockIdx.x * 128 + wn * 64;

    v8f acc[4][4];
#pragma unroll
    for (int i = 0; i < 4; ++i)
#pragma unroll
        for (int j = 0; j < 4; ++j)
            acc[i][j] = (v8f){0.f, 0.f, 0.f, 0.f, 0.f, 0.f, 0.f, 0.f};

    for (int k0 = 0; k0 < K; k0 += 32) {
        Frag a[4], b[4];
#pragma unroll
        for (int i = 0; i < 4; ++i)
            a[i] = load_frag(A + (size_t)(m0 + 16 * i) * K + k0, K);
#pragma unroll
        for (int j = 0; j < 4; ++j)
            b[j] = load_frag(W + (size_t)(n0 + 16 * j) * K + k0, K);
#pragma unroll
        for (int i = 0; i < 4; ++i)
#pragma unroll
            for (int j = 0; j < 4; ++j)
                acc[i][j] = wmma_bf16(a[i], b[j], acc[i][j]);
    }

    const int half = lane >> 4;
    const int col  = lane & 15;
#pragma unroll
    for (int j = 0; j < 4; ++j) {
        const int n = n0 + 16 * j + col;
        const float bv = bias[n];
#pragma unroll
        for (int i = 0; i < 4; ++i) {
#pragma unroll
            for (int r = 0; r < 8; ++r) {
                const int m = m0 + 16 * i + r + 8 * half;
                const float v = acc[i][j][r] + bv;
                if (mode == 2) {
                    ((float*)out)[(size_t)m * N + n] = v;
                } else {
                    const int bb = m >> 11, s = m & (SEQ - 1);
                    const int h  = n >> 6,  d = n & (DH - 1);
                    const unsigned short us = f2bf(v);
                    if (mode == 0)
                        ((unsigned short*)out)[(((size_t)bb * NH + h) * SEQ + s) * DH + d] = us;
                    else
                        ((unsigned short*)out)[(((size_t)bb * NH + h) * DH + d) * SEQ + s] = us;
                }
            }
        }
    }
}

// ---------------------------------------------------------------------------
// Flash-style causal attention.  One wave handles 16 query rows of one
// (batch, head); t streamed in chunks of 32 with online softmax.
// Q,K: [B,H,S,DH] bf16;  Vt: [B,H,DH,S] bf16;  ctx: [M_TOT, D_EMB] bf16.
// ---------------------------------------------------------------------------
__global__ __launch_bounds__(256)
void attn_kernel(const unsigned short* __restrict__ Q,
                 const unsigned short* __restrict__ Km,
                 const unsigned short* __restrict__ Vt,
                 unsigned short* __restrict__ ctx) {
    __shared__ __align__(16) unsigned short plds[8][16 * 32];

    const int wave = threadIdx.x >> 5;
    const int lane = threadIdx.x & 31;
    const int wid  = blockIdx.x * 8 + wave;
    const int st = wid & 127;          // 128 row-tiles of 16
    const int h  = (wid >> 7) & (NH - 1);
    const int b  = wid >> 11;
    const int s0 = st * 16;

    const size_t bh = (size_t)b * NH + h;
    const unsigned short* Qp = Q  + (bh * SEQ + s0) * DH;
    const unsigned short* Kp = Km + bh * SEQ * DH;
    const unsigned short* Vp = Vt + bh * DH * SEQ;

    Frag qa[2];
    qa[0] = load_frag(Qp + 0,  DH);
    qa[1] = load_frag(Qp + 32, DH);

    v8f o[4];
#pragma unroll
    for (int j = 0; j < 4; ++j)
        o[j] = (v8f){0.f, 0.f, 0.f, 0.f, 0.f, 0.f, 0.f, 0.f};
    float mrun[8], lrun[8];
#pragma unroll
    for (int r = 0; r < 8; ++r) { mrun[r] = -1e30f; lrun[r] = 0.f; }

    const int half = lane >> 4;
    const int col  = lane & 15;
    const float scale = 0.125f;  // 1/sqrt(DH)
    unsigned short* pl = plds[wave];

    for (int t0 = 0; t0 <= s0 + 15; t0 += 32) {
        // scores for t-chunk [t0, t0+32)
        v8f c[2];
#pragma unroll
        for (int jt = 0; jt < 2; ++jt) {
            const int tb = t0 + 16 * jt;
            Frag kb0 = load_frag(Kp + (size_t)tb * DH + 0,  DH);
            Frag kb1 = load_frag(Kp + (size_t)tb * DH + 32, DH);
            v8f cc = (v8f){0.f, 0.f, 0.f, 0.f, 0.f, 0.f, 0.f, 0.f};
            cc = wmma_bf16(qa[0], kb0, cc);
            cc = wmma_bf16(qa[1], kb1, cc);
            c[jt] = cc;
        }
        // scale + causal mask
#pragma unroll
        for (int jt = 0; jt < 2; ++jt) {
            const int t = t0 + 16 * jt + col;
#pragma unroll
            for (int r = 0; r < 8; ++r) {
                const int s = s0 + r + 8 * half;
                float v = c[jt][r] * scale;
                if (t > s) v = -1e30f;
                c[jt][r] = v;
            }
        }
        // row-wise max over chunk (16-lane half-wave reduction)
        float mnew[8], corr[8];
#pragma unroll
        for (int r = 0; r < 8; ++r) {
            float t = fmaxf(c[0][r], c[1][r]);
#pragma unroll
            for (int off = 8; off >= 1; off >>= 1)
                t = fmaxf(t, __shfl_xor(t, off, 32));
            mnew[r] = fmaxf(mrun[r], t);
            corr[r] = __expf(mrun[r] - mnew[r]);
            mrun[r] = mnew[r];
        }
        // p = exp(c - mnew); stash row-major bf16 into per-wave LDS tile
#pragma unroll
        for (int jt = 0; jt < 2; ++jt) {
#pragma unroll
            for (int r = 0; r < 8; ++r) {
                const float p = __expf(c[jt][r] - mnew[r]);
                c[jt][r] = p;
                pl[(r + 8 * half) * 32 + 16 * jt + col] = f2bf(p);
            }
        }
        // row sums + online rescale
#pragma unroll
        for (int r = 0; r < 8; ++r) {
            float s = c[0][r] + c[1][r];
#pragma unroll
            for (int off = 8; off >= 1; off >>= 1)
                s += __shfl_xor(s, off, 32);
            lrun[r] = lrun[r] * corr[r] + s;
        }
#pragma unroll
        for (int j = 0; j < 4; ++j)
#pragma unroll
            for (int r = 0; r < 8; ++r)
                o[j][r] *= corr[r];
        // out += P * V   (A = P from LDS, B columns = rows of Vt)
        Frag pa = load_frag(pl, 32);
#pragma unroll
        for (int j = 0; j < 4; ++j) {
            Frag vb = load_frag(Vp + (size_t)(16 * j) * SEQ + t0, SEQ);
            o[j] = wmma_bf16(pa, vb, o[j]);
        }
    }

    // normalize and emit ctx bf16 [b*SEQ+s, h*64+d]
#pragma unroll
    for (int j = 0; j < 4; ++j) {
#pragma unroll
        for (int r = 0; r < 8; ++r) {
            const int s = s0 + r + 8 * half;
            const int d = h * DH + 16 * j + col;
            ctx[(size_t)(b * SEQ + s) * D_EMB + d] = f2bf(o[j][r] / lrun[r]);
        }
    }
}

// ---------------------------------------------------------------------------
extern "C" void kernel_launch(void* const* d_in, const int* in_sizes, int n_in,
                              void* d_out, int out_size, void* d_ws, size_t ws_size,
                              hipStream_t stream) {
    const float* x  = (const float*)d_in[0];
    const float* Wq = (const float*)d_in[1];
    const float* bq = (const float*)d_in[2];
    const float* Wk = (const float*)d_in[3];
    const float* bk = (const float*)d_in[4];
    const float* Wv = (const float*)d_in[5];
    const float* bv = (const float*)d_in[6];
    const float* Wo = (const float*)d_in[7];
    const float* bo = (const float*)d_in[8];
    float* out = (float*)d_out;

    const size_t SZ_X  = (size_t)M_TOT * D_EMB * 2;   // bf16 activations
    const size_t SZ_W  = (size_t)D_EMB * D_EMB * 2;   // bf16 weights
    char* p = (char*)d_ws;
    unsigned short* xb   = (unsigned short*)(p); p += SZ_X;
    unsigned short* wqb  = (unsigned short*)(p); p += SZ_W;
    unsigned short* wkb  = (unsigned short*)(p); p += SZ_W;
    unsigned short* wvb  = (unsigned short*)(p); p += SZ_W;
    unsigned short* wob  = (unsigned short*)(p); p += SZ_W;
    unsigned short* Qb   = (unsigned short*)(p); p += SZ_X;
    unsigned short* Kb   = (unsigned short*)(p); p += SZ_X;
    unsigned short* Vtb  = (unsigned short*)(p); p += SZ_X;
    unsigned short* ctxb = (unsigned short*)(p); p += SZ_X;

    const int nx4 = M_TOT * D_EMB / 4;        // 1M
    const int nw4 = D_EMB * D_EMB / 4;        // 256K
    cvt_kernel<<<nx4 / 256, 256, 0, stream>>>(x,  xb,  nx4);
    cvt_kernel<<<nw4 / 256, 256, 0, stream>>>(Wq, wqb, nw4);
    cvt_kernel<<<nw4 / 256, 256, 0, stream>>>(Wk, wkb, nw4);
    cvt_kernel<<<nw4 / 256, 256, 0, stream>>>(Wv, wvb, nw4);
    cvt_kernel<<<nw4 / 256, 256, 0, stream>>>(Wo, wob, nw4);

    dim3 ggrid(D_EMB / 128, M_TOT / 256);     // (8,16)
    gemm_bias_kernel<<<ggrid, 256, 0, stream>>>(xb, wqb, bq, Qb,  M_TOT, D_EMB, D_EMB, 0);
    gemm_bias_kernel<<<ggrid, 256, 0, stream>>>(xb, wkb, bk, Kb,  M_TOT, D_EMB, D_EMB, 0);
    gemm_bias_kernel<<<ggrid, 256, 0, stream>>>(xb, wvb, bv, Vtb, M_TOT, D_EMB, D_EMB, 1);

    attn_kernel<<<(BATCH * NH * (SEQ / 16)) / 8, 256, 0, stream>>>(Qb, Kb, Vtb, ctxb);

    gemm_bias_kernel<<<ggrid, 256, 0, stream>>>(ctxb, wob, bo, (void*)out, M_TOT, D_EMB, D_EMB, 2);
}
